// Multi_Scale_Attention2_17523466568048
// MI455X (gfx1250) — matrix operand order repaired and checked by reading, not compiled, not run on hardware
//
#include <hip/hip_runtime.h>
#include <hip/hip_bf16.h>
#include <math.h>

// ---------------------------------------------------------------------------
// Multi_Scale_Attention2 for MI455X (gfx1250, wave32, WMMA + async LDS copies)
// B=8, C=64, H=128, W=128, 3C=192
// ---------------------------------------------------------------------------

typedef __attribute__((ext_vector_type(16))) _Float16 v16h;
typedef __attribute__((ext_vector_type(8)))  _Float16 v8h;
typedef __attribute__((ext_vector_type(4)))  _Float16 v4h;
typedef __attribute__((ext_vector_type(8)))  float    v8f;

#define Bb 8
#define Cc 64
#define Hh 128
#define Ww 128
#define C3 192

// LDS strides (halves) — odd multiples of 8 so 16-byte fragment reads spread banks
#define S_SXT  72    // sxT / xrpT rows: 64 cols + pad
#define S_PRJ  136   // x1 / xfp rows: 128 cols + pad
#define S_S1   72    // wsm / s1 rows: 64 cols + pad
#define S_CONV 40    // conv tiles: 32 cols + pad (80 B rows, 16B aligned)
#define S_S2   132   // s2 (f32)

// ---- WMMA fragment helpers (layouts per CDNA5 ISA 7.12.2, wave32) ----------

__device__ __forceinline__ v8f wmma_f16(v16h a, v16h b, v8f c) {
  return __builtin_amdgcn_wmma_f32_16x16x32_f16(false, a, false, b,
                                                (short)0, c, false, false);
}

// A 16x32 f16 fragment, row-major tile: element (m,k) = base[m*stride + k].
// Per-lane halves are two contiguous 16B runs -> two ds_load_b128.
__device__ __forceinline__ v16h load_a16x32(const _Float16* base, int stride, int lane) {
  const int m   = lane & 15;
  const int khi = (lane & 16) ? 8 : 0;
  const _Float16* row = base + m * stride + khi;
  const v8h lo = *(const v8h*)(row);
  const v8h hi = *(const v8h*)(row + 16);
  return __builtin_shufflevector(lo, hi, 0,1,2,3,4,5,6,7,8,9,10,11,12,13,14,15);
}

// B 32x16 f16 fragment from an N-major (transposed) tile:
// element (k,n) = base[n*stride + k]. Lane: n = lane&15, k = kb..kb+7 and kb+16..kb+23.
__device__ __forceinline__ v16h load_b32x16t(const _Float16* base, int stride, int lane) {
  const int n  = lane & 15;
  const int kb = (lane & 16) ? 8 : 0;
  const _Float16* row = base + n * stride + kb;
  const v8h lo = *(const v8h*)(row);
  const v8h hi = *(const v8h*)(row + 16);
  return __builtin_shufflevector(lo, hi, 0,1,2,3,4,5,6,7,8,9,10,11,12,13,14,15);
}

// Async 16B global->LDS copy (ASYNCcnt-tracked, bypasses VGPRs).
// LDS flat address low 32 bits == LDS byte offset (LDS aperture truncation).
__device__ __forceinline__ void async_copy_b128(void* lds, const void* gptr) {
  const unsigned           l = (unsigned)(size_t)lds;
  const unsigned long long g = (unsigned long long)(size_t)gptr;
  asm volatile("global_load_async_to_lds_b128 %0, %1, off"
               :: "v"(l), "v"(g) : "memory");
}
__device__ __forceinline__ void wait_async() {
  asm volatile("s_wait_asynccnt 0" ::: "memory");
}

// ---------------------------------------------------------------------------
// Kernel 1: fused projections + attention + softmax + concat(score*inputs)
// One workgroup (128 threads = 4 waves) per (b,h). Dynamic LDS (~157 KB).
// Output: cat2 in NHWC-f16 layout [b][h][w][3C] (channel-contiguous so the
// conv kernel can stage tiles with straight async B128 copies).
// ---------------------------------------------------------------------------
__global__ __launch_bounds__(128) void attn_kernel(
    const float* __restrict__ x,  const float* __restrict__ xf, const float* __restrict__ xr,
    const float* __restrict__ w_x1, const float* __restrict__ b_x1,
    const float* __restrict__ w_xf, const float* __restrict__ b_xf,
    const float* __restrict__ w_xr, const float* __restrict__ b_xr,
    _Float16* __restrict__ cat2)
{
  extern __shared__ char smem[];
  _Float16* sxT = (_Float16*)smem;             // 3 * [128 w][S_SXT] inputs, N-major
  _Float16* p0  = sxT + 3 * Ww * S_SXT;        // x1  [64 c][S_PRJ] row-major (A op)
  _Float16* p1  = p0  + Cc * S_PRJ;            // xfp [64 d][S_PRJ] row-major (B^T op)
  _Float16* p2t = p1  + Cc * S_PRJ;            // xrpT [128 w][S_SXT] N-major (B op)
  _Float16* wsm = p2t + Ww * S_SXT;            // weight [64 o][S_S1]
  _Float16* s1  = wsm + Cc * S_S1;             // score1 [64 c][S_S1]
  float*    s2  = (float*)(s1 + Cc * S_S1);    // score2 [64 c][S_S2]
  float*    bsm = s2 + Cc * S_S2;              // 64 bias

  const int tid  = threadIdx.x;
  const int lane = tid & 31;
  const int wv   = tid >> 5;           // 0..3
  const int bh   = blockIdx.x;
  const int b    = bh >> 7;
  const int h    = bh & 127;

  const float* xin[3] = {x, xf, xr};
  const float* wp[3]  = {w_x1, w_xf, w_xr};
  const float* bp[3]  = {b_x1, b_xf, b_xr};

  // ---- stage x, xf, xr slices (b,:,h,:) -> f16 LDS, N-major [w][c] ----------
  for (int p = 0; p < 3; ++p) {
    const float* src = xin[p];
    _Float16* dst = sxT + p * Ww * S_SXT;
    for (int i = tid; i < (Cc * Ww) / 4; i += 128) {
      const int c  = i >> 5;
      const int w4 = (i & 31) << 2;
      const float4 f = *(const float4*)&src[(((size_t)b * Cc + c) * Hh + h) * Ww + w4];
      dst[(w4 + 0) * S_SXT + c] = (_Float16)f.x;
      dst[(w4 + 1) * S_SXT + c] = (_Float16)f.y;
      dst[(w4 + 2) * S_SXT + c] = (_Float16)f.z;
      dst[(w4 + 3) * S_SXT + c] = (_Float16)f.w;
    }
  }
  __syncthreads();

  // ---- projections: proj_p (64x128) = W_p (64x64) @ sx_p (64x128) + b_p -----
  for (int p = 0; p < 3; ++p) {
    for (int i = tid; i < Cc * Cc; i += 128) {
      const int o = i >> 6, c = i & 63;
      wsm[o * S_S1 + c] = (_Float16)wp[p][i];
    }
    if (tid < Cc) bsm[tid] = bp[p][tid];
    __syncthreads();

    const _Float16* bt = sxT + p * Ww * S_SXT;
    const int m0 = wv * 16;
    v8f acc[8] = {};
    for (int k0 = 0; k0 < Cc; k0 += 32) {
      const v16h a = load_a16x32(wsm + m0 * S_S1 + k0, S_S1, lane);
#pragma unroll
      for (int t = 0; t < 8; ++t) {
        const v16h bf = load_b32x16t(bt + (t * 16) * S_SXT + k0, S_SXT, lane);
        acc[t] = wmma_f16(a, bf, acc[t]);
      }
    }
#pragma unroll
    for (int t = 0; t < 8; ++t)
#pragma unroll
      for (int r = 0; r < 8; ++r) {
        const int m = m0 + r + ((lane & 16) ? 8 : 0);
        const int n = t * 16 + (lane & 15);
        const _Float16 v = (_Float16)(acc[t][r] + bsm[m]);
        if (p == 0)      p0[m * S_PRJ + n] = v;       // row-major
        else if (p == 1) p1[m * S_PRJ + n] = v;       // row-major
        else             p2t[n * S_SXT + m] = v;      // N-major
      }
    __syncthreads();
  }

  // ---- score1 (64x64) = x1 (64x128) @ xfp^T (128x64) / sqrt(C) --------------
  {
    const int m0 = wv * 16;
    v8f acc[4] = {};
    for (int k0 = 0; k0 < Ww; k0 += 32) {
      const v16h a = load_a16x32(p0 + m0 * S_PRJ + k0, S_PRJ, lane);
#pragma unroll
      for (int t = 0; t < 4; ++t) {
        const v16h bf = load_b32x16t(p1 + (t * 16) * S_PRJ + k0, S_PRJ, lane);
        acc[t] = wmma_f16(a, bf, acc[t]);
      }
    }
#pragma unroll
    for (int t = 0; t < 4; ++t)
#pragma unroll
      for (int r = 0; r < 8; ++r) {
        const int m = m0 + r + ((lane & 16) ? 8 : 0);
        const int n = t * 16 + (lane & 15);
        s1[m * S_S1 + n] = (_Float16)(acc[t][r] * 0.125f);   // 1/sqrt(64)
      }
  }
  __syncthreads();

  // ---- score2 (64x128) = score1 (64x64) @ xrp (64x128) / sqrt(W) ------------
  {
    const int m0 = wv * 16;
    v8f acc[8] = {};
    for (int k0 = 0; k0 < Cc; k0 += 32) {
      const v16h a = load_a16x32(s1 + m0 * S_S1 + k0, S_S1, lane);
#pragma unroll
      for (int t = 0; t < 8; ++t) {
        const v16h bf = load_b32x16t(p2t + (t * 16) * S_SXT + k0, S_SXT, lane);
        acc[t] = wmma_f16(a, bf, acc[t]);
      }
    }
    const float inv = 0.08838834764831845f;  // 1/sqrt(128)
#pragma unroll
    for (int t = 0; t < 8; ++t)
#pragma unroll
      for (int r = 0; r < 8; ++r) {
        const int m = m0 + r + ((lane & 16) ? 8 : 0);
        const int n = t * 16 + (lane & 15);
        s2[m * S_S2 + n] = acc[t][r] * inv;
      }
  }
  __syncthreads();

  // ---- softmax over w (rows of 128) ----------------------------------------
  if (tid < Cc) {
    float* row = s2 + tid * S_S2;
    float mx = row[0];
    for (int i = 1; i < Ww; ++i) mx = fmaxf(mx, row[i]);
    float sum = 0.f;
    for (int i = 0; i < Ww; ++i) { const float e = expf(row[i] - mx); row[i] = e; sum += e; }
    const float r = 1.f / sum;
    for (int i = 0; i < Ww; ++i) row[i] *= r;
  }
  __syncthreads();

  // ---- concat(score*x, score*xf, score*xr) -> NHWC f16 cat2 -----------------
  {
    const size_t slab = (((size_t)b * Hh + h) * Ww) * C3;
    for (int j = tid; j < (Ww * C3) / 4; j += 128) {
      const int off = j << 2;          // half offset within slab, multiple of 4
      const int w   = off / C3;
      const int r4  = off % C3;        // multiple of 4, stays inside one p-block
      const int p   = r4 >> 6;
      const int c4  = r4 & 63;
      const _Float16* sp = sxT + p * Ww * S_SXT;
      v4h o;
#pragma unroll
      for (int jj = 0; jj < 4; ++jj)
        o[jj] = (_Float16)(s2[(c4 + jj) * S_S2 + w] * (float)sp[w * S_SXT + c4 + jj]);
      *(v4h*)&cat2[slab + off] = o;
    }
  }
}

// ---------------------------------------------------------------------------
// Kernel 2: 3x3 conv (192->192) as implicit GEMM, double-buffered async LDS
// staging. Tile: 64 out-channels x 64 pixels per 256-thread WG (8 waves).
// K loop: 54 chunks (9 taps x 6 chunks of 32 channels). Both operand tiles are
// contiguous 16B runs in global -> one global_load_async_to_lds_b128 per
// thread per operand per chunk; zero VGPR traffic for staging.
// ---------------------------------------------------------------------------
__global__ __launch_bounds__(256) void conv_kernel(
    const _Float16* __restrict__ cat2,   // [b][h][w][3C] f16
    const _Float16* __restrict__ wpack,  // [rs][k][c]    f16
    const float* __restrict__ scaleA, const float* __restrict__ shiftA,
    float* __restrict__ y, float* __restrict__ pool)
{
  __shared__ _Float16 As[2][64 * S_CONV];   // weights (k rows x 32 c)
  __shared__ _Float16 Bt[2][64 * S_CONV];   // input   (n rows x 32 c), N-major
  __shared__ float    pl[64];               // per-channel partial sums

  const int tid  = threadIdx.x;
  const int lane = tid & 31;
  const int wv   = tid >> 5;             // 0..7
  const int kBase = blockIdx.x * 64;     // 0,64,128
  int sid = blockIdx.y;                  // 0..2047
  const int b  = sid >> 8; sid &= 255;
  const int h  = sid >> 1;
  const int w0 = (sid & 1) * 64;

  if (tid < 64) pl[tid] = 0.f;

  const int nrow = tid >> 2;             // 0..63: tile row this thread stages
  const int q8   = (tid & 3) << 3;       // 16B quarter within the 32-half row

  const int m0 = (wv >> 1) * 16;         // 0,16,32,48
  const int n0 = (wv & 1) * 32;          // 0,32
  v8f acc[2] = {};

  auto issue = [&](int j, int buf) {
    const int rs = j / 6, c0 = (j % 6) * 32;
    const int r = rs / 3 - 1, s = rs % 3 - 1;
    const int hh = h + r, ww = w0 + nrow + s;
    // A: packed weights, always in bounds
    async_copy_b128(&As[buf][nrow * S_CONV + q8],
                    wpack + ((size_t)(rs * C3 + kBase + nrow)) * C3 + c0 + q8);
    // B: NHWC input row, zero-fill at image borders
    if (hh >= 0 && hh < Hh && ww >= 0 && ww < Ww) {
      async_copy_b128(&Bt[buf][nrow * S_CONV + q8],
                      cat2 + (((size_t)b * Hh + hh) * Ww + ww) * C3 + c0 + q8);
    } else {
      v8h z = {};
      *(v8h*)&Bt[buf][nrow * S_CONV + q8] = z;
    }
  };

  issue(0, 0);
  for (int j = 0; j < 54; ++j) {
    const int buf = j & 1;
    wait_async();            // this wave's copies for chunk j have landed
    __syncthreads();         // everyone's copies landed; prev buffer free
    if (j + 1 < 54) issue(j + 1, buf ^ 1);   // prefetch next chunk

    const v16h a = load_a16x32(As[buf] + m0 * S_CONV, S_CONV, lane);
#pragma unroll
    for (int t = 0; t < 2; ++t) {
      const v16h bf = load_b32x16t(Bt[buf] + (n0 + t * 16) * S_CONV, S_CONV, lane);
      acc[t] = wmma_f16(a, bf, acc[t]);
    }
  }
  __syncthreads();

  // epilogue: y = relu(conv*scale + shift), accumulate channel sums
#pragma unroll
  for (int t = 0; t < 2; ++t)
#pragma unroll
    for (int r8 = 0; r8 < 8; ++r8) {
      const int ml = m0 + r8 + ((lane & 16) ? 8 : 0);
      const int k  = kBase + ml;
      const int n  = n0 + t * 16 + (lane & 15);
      const float v = fmaxf(acc[t][r8] * scaleA[k] + shiftA[k], 0.f);
      y[(((size_t)b * C3 + k) * Hh + h) * Ww + (w0 + n)] = v;
      atomicAdd(&pl[ml], v);
    }
  __syncthreads();
  if (tid < 64) atomicAdd(&pool[b * C3 + kBase + tid], pl[tid]);
}

// ---------------------------------------------------------------------------
// Kernel 3a: pack conv weights (O,I,3,3) f32 -> [rs][k][c] f16 (c-contiguous)
// ---------------------------------------------------------------------------
__global__ void pack_w_kernel(const float* __restrict__ w_conv,
                              _Float16* __restrict__ wpack)
{
  const int i = blockIdx.x * blockDim.x + threadIdx.x;   // flat (k,c,rs)
  if (i < C3 * C3 * 9) {
    const int rs = i % 9;
    const int c  = (i / 9) % C3;
    const int k  = i / (9 * C3);
    wpack[((size_t)rs * C3 + k) * C3 + c] = (_Float16)w_conv[i];
  }
}

// ---------------------------------------------------------------------------
// Kernel 3b: fold bias+BN into per-channel scale/shift; zero pool accumulators
// ---------------------------------------------------------------------------
__global__ void prep_kernel(const float* __restrict__ b_conv,
                            const float* __restrict__ gamma, const float* __restrict__ beta,
                            const float* __restrict__ mean,  const float* __restrict__ var,
                            float* __restrict__ scaleA, float* __restrict__ shiftA,
                            float* __restrict__ pool)
{
  const int i = blockIdx.x * blockDim.x + threadIdx.x;
  if (i < C3) {
    const float sc = gamma[i] * rsqrtf(var[i] + 1e-4f);
    scaleA[i] = sc;
    shiftA[i] = beta[i] + (b_conv[i] - mean[i]) * sc;
  }
  if (i < Bb * C3) pool[i] = 0.f;
}

// ---------------------------------------------------------------------------
// Kernel 4: ECA gate = sigmoid(conv1d(avgpool, k=3, pad=1))
// ---------------------------------------------------------------------------
__global__ void eca_kernel(const float* __restrict__ pool,
                           const float* __restrict__ eca_w,
                           float* __restrict__ gate)
{
  __shared__ float p[C3 + 2];
  const int b = blockIdx.x, c = threadIdx.x;
  if (c == 0)  p[0]      = 0.f;
  if (c == 1)  p[C3 + 1] = 0.f;
  p[c + 1] = pool[b * C3 + c] * (1.f / (Hh * Ww));
  __syncthreads();
  const float z = eca_w[0] * p[c] + eca_w[1] * p[c + 1] + eca_w[2] * p[c + 2];
  gate[b * C3 + c] = 1.f / (1.f + expf(-z));
}

// ---------------------------------------------------------------------------
// Kernel 5: apply gate
// ---------------------------------------------------------------------------
__global__ __launch_bounds__(256) void scale_kernel(float* __restrict__ y,
                                                    const float* __restrict__ gate,
                                                    int n)
{
  const int i = blockIdx.x * blockDim.x + threadIdx.x;
  if (i < n) y[i] *= gate[i >> 14];   // 16384 = H*W pixels per (b,channel)
}

// ---------------------------------------------------------------------------

extern "C" void kernel_launch(void* const* d_in, const int* in_sizes, int n_in,
                              void* d_out, int out_size, void* d_ws, size_t ws_size,
                              hipStream_t stream) {
  const float* x      = (const float*)d_in[0];
  const float* xf     = (const float*)d_in[1];
  const float* xr     = (const float*)d_in[2];
  const float* w_x1   = (const float*)d_in[3];
  const float* b_x1   = (const float*)d_in[4];
  const float* w_xf   = (const float*)d_in[5];
  const float* b_xf   = (const float*)d_in[6];
  const float* w_xr   = (const float*)d_in[7];
  const float* b_xr   = (const float*)d_in[8];
  const float* w_conv = (const float*)d_in[9];
  const float* b_conv = (const float*)d_in[10];
  const float* gamma  = (const float*)d_in[11];
  const float* beta   = (const float*)d_in[12];
  const float* mean   = (const float*)d_in[13];
  const float* var    = (const float*)d_in[14];
  const float* eca_w  = (const float*)d_in[15];
  float* y = (float*)d_out;

  // workspace: NHWC f16 concat, packed f16 weights, pool/gate/scale/shift f32
  const size_t catElems = (size_t)Bb * C3 * Hh * Ww;       // 25,165,824
  const size_t wpElems  = (size_t)9 * C3 * C3;             // 331,776
  _Float16* cat2   = (_Float16*)d_ws;
  _Float16* wpack  = cat2 + catElems;
  float*    pool   = (float*)(wpack + wpElems);
  float*    gate   = pool + Bb * C3;
  float*    scaleA = gate + Bb * C3;
  float*    shiftA = scaleA + C3;

  // 0) fold BN constants, zero pool accumulators, pack conv weights
  prep_kernel<<<(Bb * C3 + 255) / 256, 256, 0, stream>>>(
      b_conv, gamma, beta, mean, var, scaleA, shiftA, pool);
  pack_w_kernel<<<(int)((wpElems + 255) / 256), 256, 0, stream>>>(w_conv, wpack);

  // 1) fused attention: one WG per (b,h)
  const size_t attnLds =
      ((size_t)3 * Ww * S_SXT +     // sxT
       (size_t)2 * Cc * S_PRJ +     // x1, xfp
       (size_t)Ww * S_SXT +         // xrpT
       (size_t)2 * Cc * S_S1) * 2   // wsm, s1
      + ((size_t)Cc * S_S2 + Cc) * 4;  // s2, bias
  attn_kernel<<<Bb * Hh, 128, attnLds, stream>>>(
      x, xf, xr, w_x1, b_x1, w_xf, b_xf, w_xr, b_xr, cat2);

  // 2) implicit-GEMM 3x3 conv + BN + ReLU + channel pooling (async staging)
  conv_kernel<<<dim3(3, Bb * Hh * 2), 256, 0, stream>>>(
      cat2, wpack, scaleA, shiftA, y, pool);

  // 3) ECA gate
  eca_kernel<<<Bb, C3, 0, stream>>>(pool, eca_w, gate);

  // 4) apply gate
  const int n = (int)catElems;
  scale_kernel<<<(n + 255) / 256, 256, 0, stream>>>(y, gate, n);
}
